// SkipGramNetwork_9534827397459
// MI455X (gfx1250) — matrix operand order repaired and checked by reading, not compiled, not run on hardware
//
#include <hip/hip_runtime.h>
#include <cstdint>
#include <cstddef>

// Problem sizes (fixed by the reference)
constexpr int VOC   = 50257;
constexpr int EMB   = 128;
constexpr int BATCH = 4096;

constexpr int VCHUNK     = 512;                         // v-columns per block
constexpr int NCHUNK     = (VOC + VCHUNK - 1) / VCHUNK; // 99
constexpr int STAGE_COLS = 64;                          // W2 cols staged in LDS per step
constexpr int NSTAGES    = VCHUNK / STAGE_COLS;         // 8
constexpr int COL_STRIDE = 136;                         // bf16 elems (272 B) -> conflict-free
constexpr float NEG_BIG  = -3.0e38f;                    // finite "-inf" (avoids inf-inf NaNs)

typedef __bf16 bf16x8   __attribute__((ext_vector_type(8)));
typedef __bf16 bf16x16  __attribute__((ext_vector_type(16)));
typedef float  f32x8    __attribute__((ext_vector_type(8)));

__device__ __forceinline__ f32x8 wmma_bf16(bf16x16 a, bf16x16 b, f32x8 c) {
  // D = A(16x32 bf16) * B(32x16 bf16) + C(16x16 f32)
  return __builtin_amdgcn_wmma_f32_16x16x32_bf16(false, a, false, b, (short)0, c,
                                                 false, false);
}

// A-matrix 16x32 bf16 fragment: lanes 0-15 hold K 0..15,
// lanes 16-31 hold K 16..31, row M = lane%16. Two 16B global loads.
__device__ __forceinline__ bf16x16 load_a_frag(const __bf16* __restrict__ hrow,
                                               int kt, int laneHi) {
  const int kb = kt * 32 + (laneHi ? 16 : 0);
  bf16x8 lo = *(const bf16x8*)(hrow + kb);
  bf16x8 hi = *(const bf16x8*)(hrow + kb + 8);
  return __builtin_shufflevector(lo, hi, 0,1,2,3,4,5,6,7,8,9,10,11,12,13,14,15);
}

// B-matrix 32x16 bf16 fragment: lane holds column N = lane%16; lanes 0-15 carry
// K 0..15, lanes 16-31 carry K 16..31 (16 contiguous bf16 -> one 32B LDS read).
__device__ __forceinline__ bf16x16 load_b_frag(const __bf16* scol, int kt, int laneHi) {
  const int kb = kt * 32 + (laneHi ? 16 : 0);
  return *(const bf16x16*)(scol + kb);
}

// Generic pointers to __shared__ carry the LDS byte address in their low 32 bits
// (flat->LDS mapping truncates to addr[31:0]).
__device__ __forceinline__ uint32_t lds_addr32(const void* p) {
  return (uint32_t)(uintptr_t)p;
}

// Async-copy one thread's 64B slice of a W2 tile row into LDS (no VGPR staging;
// tracked by ASYNCcnt). INST_OFFSET advances both LDS and global addresses.
__device__ __forceinline__ void stage_async_load(uint32_t ldsa,
                                                 const __bf16* __restrict__ W2bf,
                                                 int vstage, int c, int q) {
  int vc = vstage + c; if (vc >= VOC) vc = VOC - 1;   // clamp; junk cols guarded later
  uint64_t ga = (uint64_t)(uintptr_t)(W2bf + (size_t)vc * EMB) + (uint32_t)(q * 64);
  asm volatile("global_load_async_to_lds_b128 %0, %1, off"           :: "v"(ldsa), "v"(ga) : "memory");
  asm volatile("global_load_async_to_lds_b128 %0, %1, off offset:16" :: "v"(ldsa), "v"(ga) : "memory");
  asm volatile("global_load_async_to_lds_b128 %0, %1, off offset:32" :: "v"(ldsa), "v"(ga) : "memory");
  asm volatile("global_load_async_to_lds_b128 %0, %1, off offset:48" :: "v"(ldsa), "v"(ga) : "memory");
}

// ---------------- Kernel 1: H = relu(gather(W1, idx) + b1) -> bf16 ----------------
__global__ void sg_embed(const int* __restrict__ idx, const float* __restrict__ W1,
                         const float* __restrict__ b1, __bf16* __restrict__ Hbf) {
  const int b = blockIdx.x;    // BATCH blocks
  const int e = threadIdx.x;   // EMB threads
  const int id = idx[b];
  float x = W1[(size_t)e * VOC + id] + b1[e];
  x = fmaxf(x, 0.0f);
  Hbf[(size_t)b * EMB + e] = (__bf16)x;
}

// ---------------- Kernel 2: W2 f32 -> bf16 ----------------
__global__ void sg_cvt_w2(const float* __restrict__ W2, __bf16* __restrict__ W2bf, int n) {
  const int i = blockIdx.x * 256 + threadIdx.x;
  if (i < n) W2bf[i] = (__bf16)W2[i];
}

// ---------------- Kernel 3: stats GEMM (online max / sum-exp per row) ----------------
__global__ __launch_bounds__(256)
void sg_stats(const __bf16* __restrict__ Hbf, const __bf16* __restrict__ W2bf,
              const float* __restrict__ b2, float2* __restrict__ partial) {
  __shared__ __align__(16) __bf16 sW[2][STAGE_COLS * COL_STRIDE];  // double buffer
  __shared__ float2 sStat[256 * 8];

  const int tid    = threadIdx.x;
  const int lane   = tid & 31, wave = tid >> 5;     // 8 waves
  const int laneHi = lane >> 4, laneN = lane & 15;
  const int chunk  = blockIdx.x;
  const int v0     = chunk * VCHUNK;
  const int rowBase = blockIdx.y * 128;             // 128 rows per block (16/wave)

  // Resident A fragments: full K=128 for this wave's 16 rows (32 VGPRs).
  const __bf16* hrow = Hbf + (size_t)(rowBase + wave * 16 + laneN) * EMB;
  bf16x16 a0 = load_a_frag(hrow, 0, laneHi);
  bf16x16 a1 = load_a_frag(hrow, 1, laneHi);
  bf16x16 a2 = load_a_frag(hrow, 2, laneHi);
  bf16x16 a3 = load_a_frag(hrow, 3, laneHi);

  float m[8], s[8];
#pragma unroll
  for (int g = 0; g < 8; ++g) { m[g] = NEG_BIG; s[g] = 0.0f; }

  const int c = tid >> 2, q = tid & 3;              // staging roles: 64B per thread
  const uint32_t ldsSlice[2] = {
    lds_addr32(&sW[0][c * COL_STRIDE]) + (uint32_t)(q * 64),
    lds_addr32(&sW[1][c * COL_STRIDE]) + (uint32_t)(q * 64)
  };

  // Prologue: kick off stage 0.
  stage_async_load(ldsSlice[0], W2bf, v0, c, q);

  for (int st = 0; st < NSTAGES; ++st) {
    // Prefetch next stage into the other buffer, then wait for current stage.
    if (st + 1 < NSTAGES) {
      stage_async_load(ldsSlice[(st + 1) & 1], W2bf, v0 + (st + 1) * STAGE_COLS, c, q);
      asm volatile("s_wait_asynccnt 0x4" ::: "memory");   // current stage's 4 done
    } else {
      asm volatile("s_wait_asynccnt 0x0" ::: "memory");
    }
    __syncthreads();                                       // all waves' data in LDS

    const __bf16* buf = sW[st & 1];
    const int vst = v0 + st * STAGE_COLS;
#pragma unroll
    for (int nt = 0; nt < 4; ++nt) {
      const __bf16* scol = buf + (nt * 16 + laneN) * COL_STRIDE;
      f32x8 acc = {0.f, 0.f, 0.f, 0.f, 0.f, 0.f, 0.f, 0.f};
      acc = wmma_bf16(a0, load_b_frag(scol, 0, laneHi), acc);
      acc = wmma_bf16(a1, load_b_frag(scol, 1, laneHi), acc);
      acc = wmma_bf16(a2, load_b_frag(scol, 2, laneHi), acc);
      acc = wmma_bf16(a3, load_b_frag(scol, 3, laneHi), acc);
      const int v = vst + nt * 16 + laneN;
      if (v < VOC) {
        const float bias = b2[v];
#pragma unroll
        for (int g = 0; g < 8; ++g) {
          const float x  = acc[g] + bias;
          const float mo = m[g];
          const float mn = fmaxf(mo, x);
          s[g] = s[g] * __expf(mo - mn) + __expf(x - mn);
          m[g] = mn;
        }
      }
    }
    __syncthreads();   // everyone done reading buf before it is overwritten
  }

  // Cross-lane reduce: lane holds stats for rows rowBase + wave*16 + 8*laneHi + g,
  // over its column residue laneN. Merge the 16 residues per row via LDS.
#pragma unroll
  for (int g = 0; g < 8; ++g) sStat[tid * 8 + g] = make_float2(m[g], s[g]);
  __syncthreads();
  if (tid < 128) {
    const int w = tid >> 4, hb = (tid >> 3) & 1, g = tid & 7;
    float M = NEG_BIG, S = 0.0f;
#pragma unroll
    for (int l = 0; l < 16; ++l) {
      const float2 p = sStat[(w * 32 + hb * 16 + l) * 8 + g];
      const float mn = fmaxf(M, p.x);
      S = S * __expf(M - mn) + p.y * __expf(p.x - mn);
      M = mn;
    }
    partial[(size_t)(rowBase + tid) * NCHUNK + chunk] = make_float2(M, S);
  }
}

// ---------------- Kernel 4: combine per-chunk partials -> per-row offset ----------------
__global__ void sg_combine(const float2* __restrict__ partial, float* __restrict__ offs) {
  const int r = blockIdx.x * 256 + threadIdx.x;
  if (r >= BATCH) return;
  float M = NEG_BIG, S = 0.0f;
  for (int cix = 0; cix < NCHUNK; ++cix) {
    const float2 p = partial[(size_t)r * NCHUNK + cix];
    const float mn = fmaxf(M, p.x);
    S = S * __expf(M - mn) + p.y * __expf(p.x - mn);
    M = mn;
  }
  offs[r] = M + logf(S);   // logsumexp
}

// ---------------- Kernel 5: output GEMM, write log-softmax (NT stores) ----------------
__global__ __launch_bounds__(256)
void sg_out(const __bf16* __restrict__ Hbf, const __bf16* __restrict__ W2bf,
            const float* __restrict__ b2, const float* __restrict__ offs,
            float* __restrict__ out) {
  __shared__ __align__(16) __bf16 sW[2][STAGE_COLS * COL_STRIDE];  // double buffer

  const int tid    = threadIdx.x;
  const int lane   = tid & 31, wave = tid >> 5;
  const int laneHi = lane >> 4, laneN = lane & 15;
  const int chunk  = blockIdx.x;
  const int v0     = chunk * VCHUNK;
  const int rowBase = blockIdx.y * 128;

  const __bf16* hrow = Hbf + (size_t)(rowBase + wave * 16 + laneN) * EMB;
  bf16x16 a0 = load_a_frag(hrow, 0, laneHi);
  bf16x16 a1 = load_a_frag(hrow, 1, laneHi);
  bf16x16 a2 = load_a_frag(hrow, 2, laneHi);
  bf16x16 a3 = load_a_frag(hrow, 3, laneHi);

  // Hoist this lane's 8 output rows and their logsumexp offsets.
  const int rowG = rowBase + wave * 16 + laneHi * 8;   // rows rowG..rowG+7 (C layout)
  float offg[8];
#pragma unroll
  for (int g = 0; g < 8; ++g) offg[g] = offs[rowG + g];

  const int c = tid >> 2, q = tid & 3;
  const uint32_t ldsSlice[2] = {
    lds_addr32(&sW[0][c * COL_STRIDE]) + (uint32_t)(q * 64),
    lds_addr32(&sW[1][c * COL_STRIDE]) + (uint32_t)(q * 64)
  };

  stage_async_load(ldsSlice[0], W2bf, v0, c, q);

  for (int st = 0; st < NSTAGES; ++st) {
    if (st + 1 < NSTAGES) {
      stage_async_load(ldsSlice[(st + 1) & 1], W2bf, v0 + (st + 1) * STAGE_COLS, c, q);
      asm volatile("s_wait_asynccnt 0x4" ::: "memory");
    } else {
      asm volatile("s_wait_asynccnt 0x0" ::: "memory");
    }
    __syncthreads();

    const __bf16* buf = sW[st & 1];
    const int vst = v0 + st * STAGE_COLS;
#pragma unroll
    for (int nt = 0; nt < 4; ++nt) {
      const __bf16* scol = buf + (nt * 16 + laneN) * COL_STRIDE;
      f32x8 acc = {0.f, 0.f, 0.f, 0.f, 0.f, 0.f, 0.f, 0.f};
      acc = wmma_bf16(a0, load_b_frag(scol, 0, laneHi), acc);
      acc = wmma_bf16(a1, load_b_frag(scol, 1, laneHi), acc);
      acc = wmma_bf16(a2, load_b_frag(scol, 2, laneHi), acc);
      acc = wmma_bf16(a3, load_b_frag(scol, 3, laneHi), acc);
      const int v = vst + nt * 16 + laneN;
      if (v < VOC) {
        const float bias = b2[v];
#pragma unroll
        for (int g = 0; g < 8; ++g) {
          const float val = acc[g] + bias - offg[g];
          __builtin_nontemporal_store(val, out + (size_t)(rowG + g) * VOC + v);
        }
      }
    }
    __syncthreads();
  }
}

// ---------------- Launch ----------------
extern "C" void kernel_launch(void* const* d_in, const int* in_sizes, int n_in,
                              void* d_out, int out_size, void* d_ws, size_t ws_size,
                              hipStream_t stream) {
  const int*   idx = (const int*)d_in[0];
  const float* W1  = (const float*)d_in[1];
  const float* b1  = (const float*)d_in[2];
  const float* W2  = (const float*)d_in[3];
  const float* b2  = (const float*)d_in[4];
  float* out = (float*)d_out;

  // Workspace layout (all 16B-aligned): H bf16 | W2 bf16 | partial float2 | offs f32
  char* ws = (char*)d_ws;
  constexpr size_t HBF_B  = (size_t)BATCH * EMB * sizeof(__bf16);       // 1 MiB
  constexpr size_t W2BF_B = (size_t)VOC * EMB * sizeof(__bf16);         // ~12.3 MiB
  constexpr size_t PART_B = (size_t)BATCH * NCHUNK * sizeof(float2);    // ~3.1 MiB
  __bf16* Hbf     = (__bf16*)ws;
  __bf16* W2bf    = (__bf16*)(ws + HBF_B);
  float2* partial = (float2*)(ws + HBF_B + W2BF_B);
  float*  offs    = (float*)(ws + HBF_B + W2BF_B + PART_B);

  sg_embed<<<BATCH, EMB, 0, stream>>>(idx, W1, b1, Hbf);
  sg_cvt_w2<<<(VOC * EMB + 255) / 256, 256, 0, stream>>>(W2, W2bf, VOC * EMB);

  dim3 grid(NCHUNK, BATCH / 128);
  sg_stats<<<grid, 256, 0, stream>>>(Hbf, W2bf, b2, partial);
  sg_combine<<<(BATCH + 255) / 256, 256, 0, stream>>>(partial, offs);
  sg_out<<<grid, 256, 0, stream>>>(Hbf, W2bf, b2, offs, out);
}
